// TCNAdvancedLiquidNetworkModel_61538291417775
// MI455X (gfx1250) — matrix operand order repaired and checked by reading, not compiled, not run on hardware
//
#include <hip/hip_runtime.h>
#include <math.h>

// ---------------------------------------------------------------------------
// MI455X (gfx1250) implementation of TCN + 2-layer liquid RNN.
//
//  * TCN convs: cheap VALU kernels (< 2 GFLOP total).
//  * State-independent matmuls hoisted into batched WMMA GEMMs (M=32768):
//    enc = a2@proj, pre_ip0/pre_tau0/pre_gx0 = enc@W (+bias fused).
//    GEMM stages its 64-col B panel into LDS with the Tensor Data Mover
//    (tensor_load_to_lds + s_wait_tensorcnt), so the K-loop overlaps
//    LOADcnt (A prefetch) with DScnt (B fragments) around each WMMA.
//  * Sequential scan: one persistent 1024-thread workgroup (32 wave32),
//    bf16 operand copies in LDS (3 x 64KB), weights N-major bf16 in L2,
//    v_wmma_f32_16x16x32_bf16 for all 7 per-step matmuls; B fragments
//    batch-loaded 2 tiles at a time so waits are partial-count.
// ---------------------------------------------------------------------------

typedef __bf16 bf16_t;
typedef bf16_t v16bf __attribute__((ext_vector_type(16)));
typedef float  v8f   __attribute__((ext_vector_type(8)));
typedef unsigned int v4u __attribute__((ext_vector_type(4)));
typedef int v4i __attribute__((ext_vector_type(4)));
typedef int v8i __attribute__((ext_vector_type(8)));

struct alignas(16) I4 { int x, y, z, w; };

__device__ __forceinline__ bf16_t f2bf(float f) {
  union { float f; unsigned u; } a; a.f = f;
  unsigned r = a.u + 0x7FFFu + ((a.u >> 16) & 1u);   // round-to-nearest-even
  union { unsigned short s; bf16_t b; } o; o.s = (unsigned short)(r >> 16);
  return o.b;
}
__device__ __forceinline__ float bf2f(bf16_t v) {
  union { unsigned short s; bf16_t b; } i; i.b = v;
  union { unsigned u; float f; } o; o.u = ((unsigned)i.s) << 16;
  return o.f;
}
__device__ __forceinline__ float sigm(float x) { return 1.0f / (1.0f + __expf(-x)); }

// Two 16-byte chunks -> one A/B fragment (16 bf16 per lane).
__device__ __forceinline__ v16bf ldfrag(const bf16_t* p0, const bf16_t* p1) {
  union { v16bf v; I4 q[2]; } u;
  u.q[0] = *(const I4*)p0;
  u.q[1] = *(const I4*)p1;
  return u.v;
}
__device__ __forceinline__ v16bf mkfrag(I4 a, I4 b) {
  union { v16bf v; I4 q[2]; } u;
  u.q[0] = a; u.q[1] = b;
  return u.v;
}

__device__ __forceinline__ v8f wmma_bf16(v16bf a, v16bf b, v8f c) {
  return __builtin_amdgcn_wmma_f32_16x16x32_bf16(false, a, false, b, (short)0, c,
                                                 false, false);
}

#define BATCH 64
#define TSTEPS 512
#define HDIM 512
#define DT_C 0.1f
#define LN_EPS_C 1e-5f

// ---------------------------------------------------------------------------
// Tensor Data Mover: 2D tile (rows x k bf16 elements, row stride = k) from
// global into LDS. D# built per CDNA5 ISA 8.3/8.4 (count=1, type=2=image,
// data_size=1 -> 2 bytes, tile_dim0=k, tile_dim1=rows, stride0=k).
// ---------------------------------------------------------------------------
#if __has_builtin(__builtin_amdgcn_tensor_load_to_lds)
#define HAVE_TDM 1
__device__ __forceinline__ void tdm_load_2d(unsigned ldsOff, const void* gaddr,
                                            unsigned k, unsigned rows,
                                            unsigned tensorRows) {
  unsigned long long ga = (unsigned long long)(size_t)gaddr;
  v4u g0;
  g0[0] = 1u;                                      // count=1 (valid), user mode
  g0[1] = ldsOff;                                  // lds_addr (bytes)
  g0[2] = (unsigned)(ga & 0xFFFFFFFFu);            // global_addr[31:0]
  g0[3] = (unsigned)((ga >> 32) & 0x01FFFFFFu)     // global_addr[56:32]
          | (2u << 30);                            // type=2 ("image")
  v8i g1;
  g1[0] = (int)(1u << 16);                                   // data_size=2B
  g1[1] = (int)((k & 0xFFFFu) << 16);                        // tensor_dim0 lo
  g1[2] = (int)(((k >> 16) & 0xFFFFu) |
                ((tensorRows & 0xFFFFu) << 16));             // dim0 hi | dim1 lo
  g1[3] = (int)(((tensorRows >> 16) & 0xFFFFu) |
                ((k & 0xFFFFu) << 16));                      // dim1 hi | tile_dim0
  g1[4] = (int)(rows & 0xFFFFu);                             // tile_dim1, tile_dim2=0
  g1[5] = (int)k;                                            // tensor_dim0_stride lo
  g1[6] = 0;
  g1[7] = 0;
  v4i g2 = {0, 0, 0, 0};
  v4i g3 = {0, 0, 0, 0};
#if defined(__clang_major__) && (__clang_major__ >= 23)
  v8i g4 = {0, 0, 0, 0, 0, 0, 0, 0};
  __builtin_amdgcn_tensor_load_to_lds(g0, g1, g2, g3, g4, 0);
#else
  __builtin_amdgcn_tensor_load_to_lds(g0, g1, g2, g3, 0);
#endif
}
#else
#define HAVE_TDM 0
#endif

// ---------------------------------------------------------------------------
// TCN encoder (dilated convs, OIH weights), outputs bf16 activation matrix.
// ---------------------------------------------------------------------------
__global__ void conv0_k(const float* __restrict__ x, const float* __restrict__ w,
                        const float* __restrict__ b, float* __restrict__ y) {
  int idx = blockIdx.x * blockDim.x + threadIdx.x;           // over B*T
  if (idx >= BATCH * TSTEPS) return;
  int bb = idx >> 9, t = idx & 511;
  const float* xp = x + bb * TSTEPS;
  float xm1 = (t > 0) ? xp[t - 1] : 0.f;
  float x0  = xp[t];
  float xp1 = (t < TSTEPS - 1) ? xp[t + 1] : 0.f;
  for (int c = 0; c < 32; ++c) {
    float acc = b[c] + w[c * 3 + 0] * xm1 + w[c * 3 + 1] * x0 + w[c * 3 + 2] * xp1;
    y[(bb * 32 + c) * TSTEPS + t] = fmaxf(acc, 0.f);
  }
}

__global__ void conv1_k(const float* __restrict__ in, const float* __restrict__ w,
                        const float* __restrict__ b, float* __restrict__ y) {
  int idx = blockIdx.x * blockDim.x + threadIdx.x;           // (bb*64+co)*512+t
  if (idx >= BATCH * 64 * TSTEPS) return;
  int t = idx & 511, co = (idx >> 9) & 63, bb = idx >> 15;
  float acc = b[co];
  for (int ci = 0; ci < 32; ++ci) {
    const float* ip = in + (bb * 32 + ci) * TSTEPS;
    const float* wp = w + (co * 32 + ci) * 3;
    if (t >= 2)              acc += wp[0] * ip[t - 2];
    acc += wp[1] * ip[t];
    if (t < TSTEPS - 2)      acc += wp[2] * ip[t + 2];
  }
  y[idx] = fmaxf(acc, 0.f);
}

__global__ void conv2_k(const float* __restrict__ in, const float* __restrict__ w,
                        const float* __restrict__ b, bf16_t* __restrict__ a2) {
  int idx = blockIdx.x * blockDim.x + threadIdx.x;           // (bb*128+co)*512+t
  if (idx >= BATCH * 128 * TSTEPS) return;
  int t = idx & 511, co = (idx >> 9) & 127, bb = idx >> 16;
  float acc = b[co];
  for (int ci = 0; ci < 64; ++ci) {
    const float* ip = in + (bb * 64 + ci) * TSTEPS;
    const float* wp = w + (co * 64 + ci) * 3;
    if (t >= 4)              acc += wp[0] * ip[t - 4];
    acc += wp[1] * ip[t];
    if (t < TSTEPS - 4)      acc += wp[2] * ip[t + 4];
  }
  a2[((size_t)bb * TSTEPS + t) * 128 + co] = f2bf(fmaxf(acc, 0.f));
}

// ---------------------------------------------------------------------------
// Weight prep: W[K,N] f32 (rowOff rows skipped) -> Wt[N,K] bf16 (N-major).
// ---------------------------------------------------------------------------
__global__ void cvtT_k(const float* __restrict__ W, bf16_t* __restrict__ Wt,
                       int K, int N, int rowOff) {
  int idx = blockIdx.x * blockDim.x + threadIdx.x;
  if (idx >= K * N) return;
  int k = idx / N, n = idx - k * N;
  Wt[(size_t)n * K + k] = f2bf(W[(size_t)(rowOff + k) * N + n]);
}

__global__ void softplus_k(const float* __restrict__ x, float* __restrict__ y, int n) {
  int i = blockIdx.x * blockDim.x + threadIdx.x;
  if (i >= n) return;
  float v = x[i];
  y[i] = (v > 20.f) ? v : log1pf(__expf(v));
}

// ---------------------------------------------------------------------------
// Batched GEMM: C[M,N] = A[M,K] @ Wt^T (+bias), bf16 in, f32 accumulate.
// Block = 8 waves; wave tile 16x64; block tile 128x64.
// B panel (64 x K bf16) staged in LDS via TDM; A globally loaded, pipelined.
// ---------------------------------------------------------------------------
__global__ __launch_bounds__(256) void gemm_bf16_wmma(
    const bf16_t* __restrict__ A, const bf16_t* __restrict__ Wt,
    const float* __restrict__ bias, bf16_t* __restrict__ C,
    int M, int K, int N) {
  extern __shared__ char sm[];
  bf16_t* sB = (bf16_t*)sm;                      // [64][K] bf16 panel

  const int wave = threadIdx.x >> 5, lane = threadIdx.x & 31;
  const int ln16 = lane & 15, lh = lane >> 4;
  const int mtile = blockIdx.x * 8 + wave;       // 16-row tile index
  const int marow = mtile * 16 + ln16;           // A row supplied by this lane
  const int nbase = blockIdx.y * 64;

  // ---- stage B panel into LDS --------------------------------------------
#if HAVE_TDM
  if (threadIdx.x == 0) {
    tdm_load_2d((unsigned)(size_t)sB, Wt + (size_t)nbase * K,
                (unsigned)K, 64u, (unsigned)N);
    __builtin_amdgcn_s_wait_tensorcnt(0);
  }
#else
  for (int i = threadIdx.x * 8; i < 64 * K; i += 256 * 8)
    *(I4*)(sB + i) = *(const I4*)(Wt + (size_t)nbase * K + i);
#endif
  __syncthreads();

  v8f zero = {0.f, 0.f, 0.f, 0.f, 0.f, 0.f, 0.f, 0.f};
  v8f acc[4];
#pragma unroll
  for (int j = 0; j < 4; ++j) acc[j] = zero;

  // ---- pipelined K loop: A from global (LOADcnt), B from LDS (DScnt) -----
  const bf16_t* apbase = A + (size_t)marow * K + lh * 16;
  I4 a0 = *(const I4*)(apbase);
  I4 a1 = *(const I4*)(apbase + 8);
  for (int kk = 0; kk < K; kk += 32) {
    v16bf af = mkfrag(a0, a1);
    if (kk + 32 < K) {                            // prefetch next A fragment
      __builtin_prefetch(apbase + kk + 96, 0, 1); // global_prefetch_b8
      a0 = *(const I4*)(apbase + kk + 32);
      a1 = *(const I4*)(apbase + kk + 40);
    }
#pragma unroll
    for (int j = 0; j < 4; ++j) {
      const bf16_t* bp = sB + (size_t)(j * 16 + ln16) * K + kk + lh * 16;
      acc[j] = wmma_bf16(af, ldfrag(bp, bp + 8), acc[j]);
    }
  }
#pragma unroll
  for (int j = 0; j < 4; ++j) {
    int n = nbase + j * 16 + ln16;
    float bs = bias ? bias[n] : 0.f;
#pragma unroll
    for (int r = 0; r < 8; ++r) {
      int m = mtile * 16 + r + lh * 8;
      C[(size_t)m * N + n] = f2bf(acc[j][r] + bs);
    }
  }
}

// ---------------------------------------------------------------------------
// Persistent liquid scan: 1 workgroup, 32 waves, full T=512 recurrence.
// States h0,h1 live in registers (owner = WMMA C-layout); bf16 copies in LDS
// feed the A operands. Weights are N-major bf16 in global (L2 resident).
// B fragments are batch-loaded two tiles at a time so the scheduler can use
// partial-count s_wait_loadcnt instead of waiting to zero per WMMA.
// ---------------------------------------------------------------------------
__global__ __launch_bounds__(1024) void liquid_scan_kernel(
    const bf16_t* __restrict__ pre_ip0, const bf16_t* __restrict__ pre_tau0,
    const bf16_t* __restrict__ pre_gx0,
    const bf16_t* __restrict__ rec0t, const bf16_t* __restrict__ wg0ht,
    const bf16_t* __restrict__ wip1t, const bf16_t* __restrict__ wtau1t,
    const bf16_t* __restrict__ wg1xt, const bf16_t* __restrict__ rec1t,
    const bf16_t* __restrict__ wg1ht,
    const float* __restrict__ spv0, const float* __restrict__ spv1,
    const float* __restrict__ bip1, const float* __restrict__ btau1,
    const float* __restrict__ bg1,
    const float* __restrict__ ln0g, const float* __restrict__ ln0b,
    float* __restrict__ h1_out) {
  extern __shared__ char smem[];
  bf16_t* sh0 = (bf16_t*)smem;                   // [64][512] bf16
  bf16_t* sh1 = sh0 + BATCH * HDIM;
  bf16_t* su  = sh1 + BATCH * HDIM;

  const int tid = threadIdx.x;
  const int wave = tid >> 5, lane = tid & 31;
  const int ln16 = lane & 15, lh = lane >> 4;
  const int m0 = (wave & 3) * 16;                // 16 batch rows owned
  const int nbase = (wave >> 2) * 64;            // 64 cols owned (4 tiles)

  int   ncol[4];
  float spv0c[4], spv1c[4], bip1c[4], btau1c[4], bg1c[4];
#pragma unroll
  for (int j = 0; j < 4; ++j) {
    int n = nbase + j * 16 + ln16;
    ncol[j] = n;
    spv0c[j] = spv0[n]; spv1c[j] = spv1[n];
    bip1c[j] = bip1[n]; btau1c[j] = btau1[n]; bg1c[j] = bg1[n];
  }
  float g0c[16], b0c[16];                        // LN params for LN phase cols
#pragma unroll
  for (int i = 0; i < 16; ++i) { g0c[i] = ln0g[lane * 16 + i]; b0c[i] = ln0b[lane * 16 + i]; }

  float h0r[4][8], h1r[4][8];
#pragma unroll
  for (int j = 0; j < 4; ++j)
#pragma unroll
    for (int r = 0; r < 8; ++r) { h0r[j][r] = 0.f; h1r[j][r] = 0.f; }

  for (int i = tid; i < BATCH * HDIM; i += 1024) { sh0[i] = f2bf(0.f); sh1[i] = f2bf(0.f); }
  __syncthreads();

  v8f zero = {0.f, 0.f, 0.f, 0.f, 0.f, 0.f, 0.f, 0.f};

  for (int t = 0; t < TSTEPS; ++t) {
    // ---- Phase A: h0 @ rec0 and h0 @ Wg0_h on this wave's 4 tiles -------
    v8f arec[4], agat[4];
#pragma unroll
    for (int j = 0; j < 4; ++j) { arec[j] = zero; agat[j] = zero; }
    for (int kk = 0; kk < HDIM; kk += 32) {
      const bf16_t* ap = sh0 + (size_t)(m0 + ln16) * HDIM + kk + lh * 16;
      v16bf af = ldfrag(ap, ap + 8);
#pragma unroll
      for (int jj = 0; jj < 4; jj += 2) {
        const bf16_t* p0 = rec0t + (size_t)ncol[jj] * HDIM + kk + lh * 16;
        const bf16_t* p1 = wg0ht + (size_t)ncol[jj] * HDIM + kk + lh * 16;
        const bf16_t* p2 = rec0t + (size_t)ncol[jj + 1] * HDIM + kk + lh * 16;
        const bf16_t* p3 = wg0ht + (size_t)ncol[jj + 1] * HDIM + kk + lh * 16;
        v16bf f0 = ldfrag(p0, p0 + 8), f1 = ldfrag(p1, p1 + 8);
        v16bf f2 = ldfrag(p2, p2 + 8), f3 = ldfrag(p3, p3 + 8);
        arec[jj]     = wmma_bf16(af, f0, arec[jj]);
        agat[jj]     = wmma_bf16(af, f1, agat[jj]);
        arec[jj + 1] = wmma_bf16(af, f2, arec[jj + 1]);
        agat[jj + 1] = wmma_bf16(af, f3, agat[jj + 1]);
      }
    }
    __syncthreads();                             // all waves done reading sh0

    // ---- layer-0 elementwise update -------------------------------------
#pragma unroll
    for (int j = 0; j < 4; ++j) {
      int n = ncol[j];
#pragma unroll
      for (int r = 0; r < 8; ++r) {
        int m = m0 + r + lh * 8;
        size_t pidx = ((size_t)m * TSTEPS + t) * HDIM + n;
        float ip   = bf2f(pre_ip0[pidx]) + arec[j][r];
        float tau  = fmaxf(spv0c[j] * sigm(bf2f(pre_tau0[pidx])), DT_C);
        float gate = sigm(bf2f(pre_gx0[pidx]) + agat[j][r]);
        float f    = tanhf(ip);
        float h    = h0r[j][r];
        h += (-h / tau + gate * f) * DT_C;
        h = fminf(fmaxf(h, -10.f), 10.f);
        h0r[j][r] = h;
        sh0[(size_t)m * HDIM + n] = f2bf(h);
      }
    }
    __syncthreads();                             // new h0 visible

    // ---- LayerNorm(h0) -> u: wave handles 2 rows ------------------------
#pragma unroll
    for (int rr = 0; rr < 2; ++rr) {
      int row = wave * 2 + rr;
      float vals[16], s = 0.f;
#pragma unroll
      for (int i = 0; i < 16; ++i) { vals[i] = bf2f(sh0[(size_t)row * HDIM + lane * 16 + i]); s += vals[i]; }
      for (int off = 16; off; off >>= 1) s += __shfl_xor(s, off, 32);
      float mean = s * (1.0f / HDIM);
      float vs = 0.f;
#pragma unroll
      for (int i = 0; i < 16; ++i) { float d = vals[i] - mean; vs += d * d; }
      for (int off = 16; off; off >>= 1) vs += __shfl_xor(vs, off, 32);
      float inv = rsqrtf(vs * (1.0f / HDIM) + LN_EPS_C);
#pragma unroll
      for (int i = 0; i < 16; ++i)
        su[(size_t)row * HDIM + lane * 16 + i] = f2bf((vals[i] - mean) * inv * g0c[i] + b0c[i]);
    }
    __syncthreads();                             // u ready

    // ---- Phase C: u@{Wip1,Wtau1,Wg1x}, h1@{rec1,Wg1h} -------------------
    v8f aip[4], atau[4], agt[4], arc[4];
#pragma unroll
    for (int j = 0; j < 4; ++j) { aip[j] = zero; atau[j] = zero; agt[j] = zero; arc[j] = zero; }
    for (int kk = 0; kk < HDIM; kk += 32) {
      const bf16_t* ap = su + (size_t)(m0 + ln16) * HDIM + kk + lh * 16;
      v16bf uf = ldfrag(ap, ap + 8);
#pragma unroll
      for (int jj = 0; jj < 4; jj += 2) {
        const bf16_t* p0 = wip1t  + (size_t)ncol[jj] * HDIM + kk + lh * 16;
        const bf16_t* p1 = wtau1t + (size_t)ncol[jj] * HDIM + kk + lh * 16;
        const bf16_t* p2 = wg1xt  + (size_t)ncol[jj] * HDIM + kk + lh * 16;
        const bf16_t* p3 = wip1t  + (size_t)ncol[jj + 1] * HDIM + kk + lh * 16;
        const bf16_t* p4 = wtau1t + (size_t)ncol[jj + 1] * HDIM + kk + lh * 16;
        const bf16_t* p5 = wg1xt  + (size_t)ncol[jj + 1] * HDIM + kk + lh * 16;
        v16bf f0 = ldfrag(p0, p0 + 8), f1 = ldfrag(p1, p1 + 8), f2 = ldfrag(p2, p2 + 8);
        v16bf f3 = ldfrag(p3, p3 + 8), f4 = ldfrag(p4, p4 + 8), f5 = ldfrag(p5, p5 + 8);
        aip[jj]      = wmma_bf16(uf, f0, aip[jj]);
        atau[jj]     = wmma_bf16(uf, f1, atau[jj]);
        agt[jj]      = wmma_bf16(uf, f2, agt[jj]);
        aip[jj + 1]  = wmma_bf16(uf, f3, aip[jj + 1]);
        atau[jj + 1] = wmma_bf16(uf, f4, atau[jj + 1]);
        agt[jj + 1]  = wmma_bf16(uf, f5, agt[jj + 1]);
      }
    }
    for (int kk = 0; kk < HDIM; kk += 32) {
      const bf16_t* ap = sh1 + (size_t)(m0 + ln16) * HDIM + kk + lh * 16;
      v16bf hf = ldfrag(ap, ap + 8);
#pragma unroll
      for (int jj = 0; jj < 4; jj += 2) {
        const bf16_t* p0 = rec1t + (size_t)ncol[jj] * HDIM + kk + lh * 16;
        const bf16_t* p1 = wg1ht + (size_t)ncol[jj] * HDIM + kk + lh * 16;
        const bf16_t* p2 = rec1t + (size_t)ncol[jj + 1] * HDIM + kk + lh * 16;
        const bf16_t* p3 = wg1ht + (size_t)ncol[jj + 1] * HDIM + kk + lh * 16;
        v16bf f0 = ldfrag(p0, p0 + 8), f1 = ldfrag(p1, p1 + 8);
        v16bf f2 = ldfrag(p2, p2 + 8), f3 = ldfrag(p3, p3 + 8);
        arc[jj]     = wmma_bf16(hf, f0, arc[jj]);
        agt[jj]     = wmma_bf16(hf, f1, agt[jj]);
        arc[jj + 1] = wmma_bf16(hf, f2, arc[jj + 1]);
        agt[jj + 1] = wmma_bf16(hf, f3, agt[jj + 1]);
      }
    }
    __syncthreads();                             // done reading sh1/su

    // ---- layer-1 elementwise update -------------------------------------
#pragma unroll
    for (int j = 0; j < 4; ++j) {
      int n = ncol[j];
#pragma unroll
      for (int r = 0; r < 8; ++r) {
        int m = m0 + r + lh * 8;
        float f    = tanhf(aip[j][r] + bip1c[j] + arc[j][r]);
        float tau  = fmaxf(spv1c[j] * sigm(atau[j][r] + btau1c[j]), DT_C);
        float gate = sigm(agt[j][r] + bg1c[j]);
        float h    = h1r[j][r];
        h += (-h / tau + gate * f) * DT_C;
        h = fminf(fmaxf(h, -10.f), 10.f);
        h1r[j][r] = h;
        sh1[(size_t)m * HDIM + n] = f2bf(h);
      }
    }
    __syncthreads();                             // h1 visible for next step
  }

#pragma unroll
  for (int j = 0; j < 4; ++j)
#pragma unroll
    for (int r = 0; r < 8; ++r)
      h1_out[(size_t)(m0 + r + lh * 8) * HDIM + ncol[j]] = h1r[j][r];
}

// ---------------------------------------------------------------------------
// Final LN(h1) @ fc_w + fc_b : one block per batch row.
// ---------------------------------------------------------------------------
__global__ void final_ln_fc(const float* __restrict__ h1, const float* __restrict__ lng,
                            const float* __restrict__ lnb, const float* __restrict__ fcw,
                            const float* __restrict__ fcb, float* __restrict__ out) {
  __shared__ float red[256];
  int b = blockIdx.x, tid = threadIdx.x;
  float v0 = h1[(size_t)b * HDIM + tid];
  float v1 = h1[(size_t)b * HDIM + 256 + tid];
  red[tid] = v0 + v1;
  __syncthreads();
  for (int o = 128; o; o >>= 1) { if (tid < o) red[tid] += red[tid + o]; __syncthreads(); }
  float mean = red[0] * (1.0f / HDIM);
  __syncthreads();
  float d0 = v0 - mean, d1 = v1 - mean;
  red[tid] = d0 * d0 + d1 * d1;
  __syncthreads();
  for (int o = 128; o; o >>= 1) { if (tid < o) red[tid] += red[tid + o]; __syncthreads(); }
  float inv = rsqrtf(red[0] * (1.0f / HDIM) + LN_EPS_C);
  __syncthreads();
  float u0 = d0 * inv * lng[tid] + lnb[tid];
  float u1 = d1 * inv * lng[256 + tid] + lnb[256 + tid];
  red[tid] = u0 * fcw[tid] + u1 * fcw[256 + tid];
  __syncthreads();
  for (int o = 128; o; o >>= 1) { if (tid < o) red[tid] += red[tid + o]; __syncthreads(); }
  if (tid == 0) out[b] = red[0] + fcb[0];
}

// ---------------------------------------------------------------------------
extern "C" void kernel_launch(void* const* d_in, const int* in_sizes, int n_in,
                              void* d_out, int out_size, void* d_ws, size_t ws_size,
                              hipStream_t stream) {
  (void)in_sizes; (void)n_in; (void)out_size; (void)ws_size;

  const float* x     = (const float*)d_in[0];
  const float* c0w   = (const float*)d_in[1];
  const float* c0b   = (const float*)d_in[2];
  const float* c1w   = (const float*)d_in[3];
  const float* c1b   = (const float*)d_in[4];
  const float* c2w   = (const float*)d_in[5];
  const float* c2b   = (const float*)d_in[6];
  const float* projw = (const float*)d_in[7];
  const float* projb = (const float*)d_in[8];
  const float* wip0  = (const float*)d_in[9];
  const float* bip0  = (const float*)d_in[10];
  const float* tb0   = (const float*)d_in[11];
  const float* wtau0 = (const float*)d_in[12];
  const float* btau0 = (const float*)d_in[13];
  const float* rec0  = (const float*)d_in[14];
  const float* wg0   = (const float*)d_in[15];
  const float* bg0   = (const float*)d_in[16];
  const float* ln0g  = (const float*)d_in[17];
  const float* ln0b  = (const float*)d_in[18];
  const float* wip1  = (const float*)d_in[19];
  const float* bip1  = (const float*)d_in[20];
  const float* tb1   = (const float*)d_in[21];
  const float* wtau1 = (const float*)d_in[22];
  const float* btau1 = (const float*)d_in[23];
  const float* rec1  = (const float*)d_in[24];
  const float* wg1   = (const float*)d_in[25];
  const float* bg1   = (const float*)d_in[26];
  const float* ln1g  = (const float*)d_in[27];
  const float* ln1b  = (const float*)d_in[28];
  const float* fcw   = (const float*)d_in[29];
  const float* fcb   = (const float*)d_in[30];

  // workspace layout
  char* ws = (char*)d_ws;
  size_t off = 0;
  auto alloc = [&](size_t bytes) { char* p = ws + off; off += (bytes + 255) & ~(size_t)255; return p; };
  const size_t BT = (size_t)BATCH * TSTEPS;              // 32768
  float*  t0      = (float*)alloc(BT * 32 * 4);
  float*  t1      = (float*)alloc(BT * 64 * 4);
  bf16_t* a2      = (bf16_t*)alloc(BT * 128 * 2);
  bf16_t* enc     = (bf16_t*)alloc(BT * HDIM * 2);
  bf16_t* preip   = (bf16_t*)alloc(BT * HDIM * 2);
  bf16_t* pretau  = (bf16_t*)alloc(BT * HDIM * 2);
  bf16_t* pregx   = (bf16_t*)alloc(BT * HDIM * 2);
  bf16_t* projwt  = (bf16_t*)alloc((size_t)128 * HDIM * 2);
  bf16_t* wip0t   = (bf16_t*)alloc((size_t)HDIM * HDIM * 2);
  bf16_t* wtau0t  = (bf16_t*)alloc((size_t)HDIM * HDIM * 2);
  bf16_t* wg0xt   = (bf16_t*)alloc((size_t)HDIM * HDIM * 2);
  bf16_t* rec0t   = (bf16_t*)alloc((size_t)HDIM * HDIM * 2);
  bf16_t* wg0ht   = (bf16_t*)alloc((size_t)HDIM * HDIM * 2);
  bf16_t* wip1t   = (bf16_t*)alloc((size_t)HDIM * HDIM * 2);
  bf16_t* wtau1t  = (bf16_t*)alloc((size_t)HDIM * HDIM * 2);
  bf16_t* wg1xt   = (bf16_t*)alloc((size_t)HDIM * HDIM * 2);
  bf16_t* rec1t   = (bf16_t*)alloc((size_t)HDIM * HDIM * 2);
  bf16_t* wg1ht   = (bf16_t*)alloc((size_t)HDIM * HDIM * 2);
  float*  spv0    = (float*)alloc(HDIM * 4);
  float*  spv1    = (float*)alloc(HDIM * 4);
  float*  h1out   = (float*)alloc((size_t)BATCH * HDIM * 4);

  const int TPB = 256;
  auto blocks = [&](size_t n) { return (unsigned)((n + TPB - 1) / TPB); };

  // TCN
  conv0_k<<<blocks(BT), TPB, 0, stream>>>(x, c0w, c0b, t0);
  conv1_k<<<blocks(BT * 64), TPB, 0, stream>>>(t0, c1w, c1b, t1);
  conv2_k<<<blocks(BT * 128), TPB, 0, stream>>>(t1, c2w, c2b, a2);

  // Weight prep (transpose to N-major bf16)
  cvtT_k<<<blocks((size_t)128 * HDIM), TPB, 0, stream>>>(projw, projwt, 128, HDIM, 0);
  cvtT_k<<<blocks((size_t)HDIM * HDIM), TPB, 0, stream>>>(wip0,  wip0t,  HDIM, HDIM, 0);
  cvtT_k<<<blocks((size_t)HDIM * HDIM), TPB, 0, stream>>>(wtau0, wtau0t, HDIM, HDIM, 0);
  cvtT_k<<<blocks((size_t)HDIM * HDIM), TPB, 0, stream>>>(wg0,   wg0xt,  HDIM, HDIM, 0);
  cvtT_k<<<blocks((size_t)HDIM * HDIM), TPB, 0, stream>>>(rec0,  rec0t,  HDIM, HDIM, 0);
  cvtT_k<<<blocks((size_t)HDIM * HDIM), TPB, 0, stream>>>(wg0,   wg0ht,  HDIM, HDIM, HDIM);
  cvtT_k<<<blocks((size_t)HDIM * HDIM), TPB, 0, stream>>>(wip1,  wip1t,  HDIM, HDIM, 0);
  cvtT_k<<<blocks((size_t)HDIM * HDIM), TPB, 0, stream>>>(wtau1, wtau1t, HDIM, HDIM, 0);
  cvtT_k<<<blocks((size_t)HDIM * HDIM), TPB, 0, stream>>>(wg1,   wg1xt,  HDIM, HDIM, 0);
  cvtT_k<<<blocks((size_t)HDIM * HDIM), TPB, 0, stream>>>(rec1,  rec1t,  HDIM, HDIM, 0);
  cvtT_k<<<blocks((size_t)HDIM * HDIM), TPB, 0, stream>>>(wg1,   wg1ht,  HDIM, HDIM, HDIM);
  softplus_k<<<blocks(HDIM), TPB, 0, stream>>>(tb0, spv0, HDIM);
  softplus_k<<<blocks(HDIM), TPB, 0, stream>>>(tb1, spv1, HDIM);

  // Batched WMMA GEMMs (M = 32768); LDS = 64 x K bf16 B panel
  dim3 g((unsigned)(BT / 128), (unsigned)(HDIM / 64));
  gemm_bf16_wmma<<<g, 256, (size_t)64 * 128  * 2, stream>>>(a2,  projwt, projb, enc,    (int)BT, 128,  HDIM);
  gemm_bf16_wmma<<<g, 256, (size_t)64 * HDIM * 2, stream>>>(enc, wip0t,  bip0,  preip,  (int)BT, HDIM, HDIM);
  gemm_bf16_wmma<<<g, 256, (size_t)64 * HDIM * 2, stream>>>(enc, wtau0t, btau0, pretau, (int)BT, HDIM, HDIM);
  gemm_bf16_wmma<<<g, 256, (size_t)64 * HDIM * 2, stream>>>(enc, wg0xt,  bg0,   pregx,  (int)BT, HDIM, HDIM);

  // Sequential liquid scan (1 WGP, 32 waves, 192 KB LDS)
  size_t smemBytes = (size_t)3 * BATCH * HDIM * sizeof(bf16_t);
  liquid_scan_kernel<<<1, 1024, smemBytes, stream>>>(
      preip, pretau, pregx, rec0t, wg0ht, wip1t, wtau1t, wg1xt, rec1t, wg1ht,
      spv0, spv1, bip1, btau1, bg1, ln0g, ln0b, h1out);

  // Final LN + FC
  final_ln_fc<<<BATCH, 256, 0, stream>>>(h1out, ln1g, ln1b, fcw, fcb, (float*)d_out);
}
